// DeepSeekV2NormalMoeLayer_27805618275266
// MI455X (gfx1250) — compile-verified, matrix operand order checked by reading, not run on hardware
//
#include <hip/hip_runtime.h>
#include <hip/hip_bf16.h>
#include <stdint.h>

// Problem constants (match reference)
#define T_TOK 2048
#define HDIM  2048
#define NEXP  8
#define IDIM  1408
#define TOPK  2

// GEMM tiling
#define BM 32
#define BN 128
#define BK 32
#define KPAD 8            // LDS K-stride = 40 bf16 = 80B (16B multiple)
#define KSTR (BK + KPAD)

typedef __bf16 bf16_t;
typedef __attribute__((ext_vector_type(16))) __bf16 v16bf;
typedef __attribute__((ext_vector_type(8)))  float  v8f;

// ---------- helpers ----------
__device__ __forceinline__ unsigned short bfbits(float f) {
    unsigned u = __float_as_uint(f);
    u += 0x7fffu + ((u >> 16) & 1u);          // round-to-nearest-even
    return (unsigned short)(u >> 16);
}
__device__ __forceinline__ bf16_t f2bf(float f) {
    unsigned short s = bfbits(f);
    return __builtin_bit_cast(bf16_t, s);
}
__device__ __forceinline__ unsigned pack2(float lo, float hi) {
    return (unsigned)bfbits(lo) | ((unsigned)bfbits(hi) << 16);
}
// load 8 f32 from a K-column (stride elements apart), pack to 8 bf16
__device__ __forceinline__ uint4 load8col(const float* __restrict__ p, int stride) {
    float v0 = p[0];
    float v1 = p[(size_t)stride];
    float v2 = p[(size_t)stride * 2];
    float v3 = p[(size_t)stride * 3];
    float v4 = p[(size_t)stride * 4];
    float v5 = p[(size_t)stride * 5];
    float v6 = p[(size_t)stride * 6];
    float v7 = p[(size_t)stride * 7];
    uint4 r;
    r.x = pack2(v0, v1); r.y = pack2(v2, v3);
    r.z = pack2(v4, v5); r.w = pack2(v6, v7);
    return r;
}
// Fragment loader: source tile stored K-major per row (row = M for A, row = N for B),
// row stride KSTR. ISA 7.12.2 wave32 16-bit layout: lane<16 -> row=lane, K in
// {0..7}u{16..23}; lane>=16 -> row=lane-16, K in {8..15}u{24..31}.
__device__ __forceinline__ v16bf frag_ld(const bf16_t* __restrict__ base, int row, int kb) {
    v16bf f;
    const bf16_t* p = base + row * KSTR + kb;
#pragma unroll
    for (int j = 0; j < 8; ++j) { f[j] = p[j]; f[8 + j] = p[16 + j]; }
    return f;
}
// Async global->LDS copy of 8 bytes (4 bf16) per lane; tracked by ASYNCcnt.
__device__ __forceinline__ void async_cp_b64(unsigned lds_addr, const void* gaddr) {
    unsigned long long ga = (unsigned long long)(size_t)gaddr;
    asm volatile("global_load_async_to_lds_b64 %0, %1, off" :: "v"(lds_addr), "v"(ga) : "memory");
}
__device__ __forceinline__ void async_wait0() {
    asm volatile("s_wait_asynccnt 0x0" ::: "memory");
}

// ---------- kernel 0: zero output + expert counts ----------
__global__ __launch_bounds__(256) void k_zero(float* __restrict__ out, int* __restrict__ counts) {
    size_t i = (size_t)blockIdx.x * blockDim.x + threadIdx.x;
    size_t n4 = (size_t)T_TOK * HDIM / 4;
    if (i < n4) ((float4*)out)[i] = make_float4(0.f, 0.f, 0.f, 0.f);
    if (blockIdx.x == 0 && threadIdx.x < NEXP) counts[threadIdx.x] = 0;
}

// ---------- kernel 1: cast hidden_states f32 -> bf16 ----------
__global__ __launch_bounds__(256) void k_cast_x(const float* __restrict__ x, bf16_t* __restrict__ xb) {
    size_t i = (size_t)blockIdx.x * blockDim.x + threadIdx.x;  // one chunk of 4
    size_t n4 = (size_t)T_TOK * HDIM / 4;
    if (i >= n4) return;
    float4 v = ((const float4*)x)[i];
    uint2 p;
    p.x = pack2(v.x, v.y);
    p.y = pack2(v.z, v.w);
    ((uint2*)xb)[i] = p;
}

// ---------- kernel 2: router (one wave per token) ----------
__global__ __launch_bounds__(256) void k_router(const float* __restrict__ hs,
                                                const float* __restrict__ gw,
                                                int* __restrict__ topi, float* __restrict__ topw,
                                                int* __restrict__ counts) {
    int gwave = (int)((blockIdx.x * blockDim.x + threadIdx.x) >> 5);
    int lane  = threadIdx.x & 31;
    if (gwave >= T_TOK) return;
    const float* x = hs + (size_t)gwave * HDIM;
    float acc[NEXP];
#pragma unroll
    for (int e = 0; e < NEXP; ++e) acc[e] = 0.f;
    for (int h = lane; h < HDIM; h += 32) {
        float xv = x[h];
#pragma unroll
        for (int e = 0; e < NEXP; ++e) acc[e] += xv * gw[e * HDIM + h];
    }
#pragma unroll
    for (int e = 0; e < NEXP; ++e) {
        float v = acc[e];
#pragma unroll
        for (int o = 16; o > 0; o >>= 1) v += __shfl_xor(v, o, 32);
        acc[e] = v;
    }
    if (lane == 0) {
        // top-2 of logits == top-2 of softmax; renormalized top-k softmax
        // equals softmax over the two selected logits.
        float b1 = -3.4e38f; int i1 = 0;
#pragma unroll
        for (int e = 0; e < NEXP; ++e) if (acc[e] > b1) { b1 = acc[e]; i1 = e; }
        float b2 = -3.4e38f; int i2 = 0;
#pragma unroll
        for (int e = 0; e < NEXP; ++e) if (e != i1 && acc[e] > b2) { b2 = acc[e]; i2 = e; }
        float r  = __expf(b2 - b1);
        float w1 = 1.f / (1.f + r);
        float w2 = r * w1;
        topi[gwave * TOPK + 0] = i1;
        topi[gwave * TOPK + 1] = i2;
        topw[gwave * TOPK + 0] = w1;
        topw[gwave * TOPK + 1] = w2;
        atomicAdd(&counts[i1], 1);
        atomicAdd(&counts[i2], 1);
    }
}

// ---------- kernel 3: tiny exclusive scan ----------
__global__ void k_scan(const int* __restrict__ counts, int* __restrict__ offs, int* __restrict__ fill) {
    if (threadIdx.x == 0 && blockIdx.x == 0) {
        int s = 0;
        for (int e = 0; e < NEXP; ++e) { offs[e] = s; fill[e] = s; s += counts[e]; }
        offs[NEXP] = s;
    }
}

// ---------- kernel 4: scatter tokens into expert-sorted order ----------
__global__ __launch_bounds__(256) void k_scatter(const int* __restrict__ topi, const float* __restrict__ topw,
                                                 int* __restrict__ fill,
                                                 int* __restrict__ ptok, float* __restrict__ pw) {
    int t = (int)(blockIdx.x * blockDim.x + threadIdx.x);
    if (t >= T_TOK) return;
#pragma unroll
    for (int k = 0; k < TOPK; ++k) {
        int e = topi[t * TOPK + k];
        int pos = atomicAdd(&fill[e], 1);
        ptok[pos] = t;
        pw[pos]   = topw[t * TOPK + k];
    }
}

// ---------- kernel 5: grouped GEMM #1  act = silu(X*Wg) * (X*Wu) ----------
__global__ __launch_bounds__(256) void k_ffn1(const bf16_t* __restrict__ Xbf,
                                              const float* __restrict__ w_gate,
                                              const float* __restrict__ w_up,
                                              const int* __restrict__ offs,
                                              const int* __restrict__ ptok,
                                              bf16_t* __restrict__ Abf) {
    const int e  = blockIdx.z;
    const int n0 = blockIdx.x * BN;
    const int rs = offs[e], re = offs[e + 1];
    const int row0 = rs + blockIdx.y * BM;
    if (row0 >= re) return;

    __shared__ bf16_t sX[BM * KSTR];   // [M][K]  row-major-K
    __shared__ bf16_t sG[BN * KSTR];   // [N][K]  transposed during staging
    __shared__ bf16_t sU[BN * KSTR];

    const int tid  = threadIdx.x;
    const int lane = tid & 31;
    const int wave = tid >> 5;
    const int wm = (wave >> 2) * 16;   // 0 or 16
    const int wn = (wave & 3) * 32;    // two 16-col fragments per wave

    v8f accg0 = {}, accg1 = {}, accu0 = {}, accu1 = {};

    // X-tile loader role: 4 bf16 (8B) per thread -> async copy to LDS
    const int xr = tid >> 3;           // 0..31 rows
    const int xc = (tid & 7) * 4;
    int tok = -1;
    if (row0 + xr < re) tok = ptok[row0 + xr];
    const unsigned xlds = (unsigned)(size_t)&sX[xr * KSTR + xc];
    const bf16_t* xgp = (tok >= 0) ? (Xbf + (size_t)tok * HDIM + xc) : (const bf16_t*)0;
    if (tok < 0) *(uint2*)&sX[xr * KSTR + xc] = make_uint2(0u, 0u);  // OOB rows stay zero
    // W-tile loader role: one K-column of 8, two passes (K and K+16)
    const int cn = tid & 127;          // 0..127 over N
    const int ck = (tid >> 7) * 8;     // 0 or 8
    const float* Wg = w_gate + (size_t)e * HDIM * IDIM;
    const float* Wu = w_up   + (size_t)e * HDIM * IDIM;

    const int lr = lane & 15;
    const int kb = (lane >> 4) * 8;

    for (int k0 = 0; k0 < HDIM; k0 += BK) {
        // stage X tile via async copy engine (ASYNCcnt)
        if (tok >= 0) async_cp_b64(xlds, xgp + k0);
        // stage Wg / Wu tiles transposed: LDS[n][k], f32 -> bf16 in registers
#pragma unroll
        for (int pass = 0; pass < 2; ++pass) {
            const int kk = ck + pass * 16;
            const float* pg = Wg + (size_t)(k0 + kk) * IDIM + n0 + cn;
            const float* pu = Wu + (size_t)(k0 + kk) * IDIM + n0 + cn;
            if (pass == 0 && k0 + BK < HDIM) {
                __builtin_prefetch(pg + (size_t)BK * IDIM, 0, 1);
                __builtin_prefetch(pu + (size_t)BK * IDIM, 0, 1);
            }
            *(uint4*)&sG[cn * KSTR + kk] = load8col(pg, IDIM);
            *(uint4*)&sU[cn * KSTR + kk] = load8col(pu, IDIM);
        }
        async_wait0();
        __syncthreads();

        v16bf af  = frag_ld(sX, wm + lr, kb);
        v16bf bg0 = frag_ld(sG, wn + lr,      kb);
        v16bf bg1 = frag_ld(sG, wn + 16 + lr, kb);
        v16bf bu0 = frag_ld(sU, wn + lr,      kb);
        v16bf bu1 = frag_ld(sU, wn + 16 + lr, kb);
        accg0 = __builtin_amdgcn_wmma_f32_16x16x32_bf16(false, af, false, bg0, (short)0, accg0, false, false);
        accg1 = __builtin_amdgcn_wmma_f32_16x16x32_bf16(false, af, false, bg1, (short)0, accg1, false, false);
        accu0 = __builtin_amdgcn_wmma_f32_16x16x32_bf16(false, af, false, bu0, (short)0, accu0, false, false);
        accu1 = __builtin_amdgcn_wmma_f32_16x16x32_bf16(false, af, false, bu1, (short)0, accu1, false, false);
        __syncthreads();
    }

    // epilogue: silu(g)*u -> bf16 rows of Abf (permuted row index space)
    const int hb = lane >> 4;
#pragma unroll
    for (int cg = 0; cg < 2; ++cg) {
        const int col = n0 + wn + cg * 16 + lr;
        v8f g8 = cg ? accg1 : accg0;
        v8f u8 = cg ? accu1 : accu0;
#pragma unroll
        for (int r = 0; r < 8; ++r) {
            int prow = row0 + wm + r + hb * 8;
            if (prow < re) {
                float g = g8[r], u = u8[r];
                float s = g * __builtin_amdgcn_rcpf(1.f + __expf(-g));  // fast silu
                Abf[(size_t)prow * IDIM + col] = f2bf(s * u);
            }
        }
    }
}

// ---------- kernel 6: grouped GEMM #2  out[t] += w * (act * Wd) ----------
__global__ __launch_bounds__(256) void k_ffn2(const bf16_t* __restrict__ Abf,
                                              const float* __restrict__ w_down,
                                              const int* __restrict__ offs,
                                              const int* __restrict__ ptok,
                                              const float* __restrict__ pw,
                                              float* __restrict__ out) {
    const int e  = blockIdx.z;
    const int n0 = blockIdx.x * BN;    // over HDIM
    const int rs = offs[e], re = offs[e + 1];
    const int row0 = rs + blockIdx.y * BM;
    if (row0 >= re) return;

    __shared__ bf16_t sA[BM * KSTR];
    __shared__ bf16_t sW[BN * KSTR];

    const int tid  = threadIdx.x;
    const int lane = tid & 31;
    const int wave = tid >> 5;
    const int wm = (wave >> 2) * 16;
    const int wn = (wave & 3) * 32;

    v8f acc0 = {}, acc1 = {};

    const int xr = tid >> 3;
    const int xc = (tid & 7) * 4;
    const bool rowok = (row0 + xr < re);
    const unsigned alds = (unsigned)(size_t)&sA[xr * KSTR + xc];
    const bf16_t* agp = rowok ? (Abf + (size_t)(row0 + xr) * IDIM + xc) : (const bf16_t*)0;
    if (!rowok) *(uint2*)&sA[xr * KSTR + xc] = make_uint2(0u, 0u);
    const int cn = tid & 127;
    const int ck = (tid >> 7) * 8;
    const float* Wd = w_down + (size_t)e * IDIM * HDIM;

    const int lr = lane & 15;
    const int kb = (lane >> 4) * 8;

    for (int k0 = 0; k0 < IDIM; k0 += BK) {
        if (rowok) async_cp_b64(alds, agp + k0);
#pragma unroll
        for (int pass = 0; pass < 2; ++pass) {
            const int kk = ck + pass * 16;
            const float* pd = Wd + (size_t)(k0 + kk) * HDIM + n0 + cn;
            if (pass == 0 && k0 + BK < IDIM) __builtin_prefetch(pd + (size_t)BK * HDIM, 0, 1);
            *(uint4*)&sW[cn * KSTR + kk] = load8col(pd, HDIM);
        }
        async_wait0();
        __syncthreads();

        v16bf af  = frag_ld(sA, wm + lr, kb);
        v16bf bw0 = frag_ld(sW, wn + lr,      kb);
        v16bf bw1 = frag_ld(sW, wn + 16 + lr, kb);
        acc0 = __builtin_amdgcn_wmma_f32_16x16x32_bf16(false, af, false, bw0, (short)0, acc0, false, false);
        acc1 = __builtin_amdgcn_wmma_f32_16x16x32_bf16(false, af, false, bw1, (short)0, acc1, false, false);
        __syncthreads();
    }

    // weighted scatter-add (exactly TOPK=2 commutative adds per out element -> deterministic)
    const int hb = lane >> 4;
#pragma unroll
    for (int cg = 0; cg < 2; ++cg) {
        const int col = n0 + wn + cg * 16 + lr;
        v8f a8 = cg ? acc1 : acc0;
#pragma unroll
        for (int r = 0; r < 8; ++r) {
            int prow = row0 + wm + r + hb * 8;
            if (prow < re) {
                int   t = ptok[prow];
                float w = pw[prow];
                atomicAdd(&out[(size_t)t * HDIM + col], w * a8[r]);
            }
        }
    }
}

// ---------- host launcher ----------
extern "C" void kernel_launch(void* const* d_in, const int* in_sizes, int n_in,
                              void* d_out, int out_size, void* d_ws, size_t ws_size,
                              hipStream_t stream) {
    (void)in_sizes; (void)n_in; (void)out_size; (void)ws_size;
    const float* hs     = (const float*)d_in[0];   // [T,H]
    const float* gate_w = (const float*)d_in[1];   // [E,H]
    const float* w_gate = (const float*)d_in[2];   // [E,H,I]
    const float* w_up   = (const float*)d_in[3];   // [E,H,I]
    const float* w_down = (const float*)d_in[4];   // [E,I,H]
    float* out = (float*)d_out;                    // [T,H]

    // carve workspace
    char* p = (char*)d_ws;
    auto take = [&p](size_t bytes) { char* r = p; p += (bytes + 255) & ~(size_t)255; return r; };
    bf16_t* Xbf   = (bf16_t*)take((size_t)T_TOK * HDIM * sizeof(bf16_t));        // 8 MB
    bf16_t* Abf   = (bf16_t*)take((size_t)T_TOK * TOPK * IDIM * sizeof(bf16_t)); // 11.5 MB
    int*    topi  = (int*)   take((size_t)T_TOK * TOPK * sizeof(int));
    float*  topw  = (float*) take((size_t)T_TOK * TOPK * sizeof(float));
    int*    cnts  = (int*)   take(NEXP * sizeof(int));
    int*    offs  = (int*)   take((NEXP + 1) * sizeof(int));
    int*    fill  = (int*)   take(NEXP * sizeof(int));
    int*    ptok  = (int*)   take((size_t)T_TOK * TOPK * sizeof(int));
    float*  pwgt  = (float*) take((size_t)T_TOK * TOPK * sizeof(float));

    const int n4blocks = (int)(((size_t)T_TOK * HDIM / 4 + 255) / 256);

    k_zero   <<<n4blocks, 256, 0, stream>>>(out, cnts);
    k_cast_x <<<n4blocks, 256, 0, stream>>>(hs, Xbf);
    k_router <<<(T_TOK * 32) / 256, 256, 0, stream>>>(hs, gate_w, topi, topw, cnts);
    k_scan   <<<1, 32, 0, stream>>>(cnts, offs, fill);
    k_scatter<<<T_TOK / 256, 256, 0, stream>>>(topi, topw, fill, ptok, pwgt);

    dim3 g1(IDIM / BN, T_TOK / BM, NEXP);   // (11, 64, 8); blocks beyond an expert's rows exit early
    k_ffn1<<<g1, 256, 0, stream>>>(Xbf, w_gate, w_up, offs, ptok, Abf);

    dim3 g2(HDIM / BN, T_TOK / BM, NEXP);   // (16, 64, 8)
    k_ffn2<<<g2, 256, 0, stream>>>(Abf, w_down, offs, ptok, pwgt, out);
}
